// FoldingNetEncoder_85169201479758
// MI455X (gfx1250) — matrix operand order repaired and checked by reading, not compiled, not run on hardware
//
#include <hip/hip_runtime.h>

typedef __attribute__((ext_vector_type(16))) _Float16 v16h;
typedef __attribute__((ext_vector_type(8)))  _Float16 v8h;
typedef __attribute__((ext_vector_type(8)))  float    v8f;

#define BB   8
#define NP   2048
#define NPTS (BB * NP)   // 16384
#define KNN  16

// ---------------------------------------------------------------------------
// kNN: one wave per point. Positions + per-row d2 cached in LDS.
// 16 rounds of (lane scan + butterfly argmin + mark) = smallest-16 selection.
// ---------------------------------------------------------------------------
__global__ void knn_kernel(const float* __restrict__ x, int* __restrict__ knn)
{
    __shared__ float px[NP];
    __shared__ float py[NP];
    __shared__ float d2s[4][NP];

    const int tid = threadIdx.x;           // 128 threads = 4 waves
    const int b   = blockIdx.x >> 9;       // 512 groups of 4 rows per batch
    const int n0  = (blockIdx.x & 511) * 4;

    const float* xb = x + (size_t)b * NP * 10;
    for (int i = tid; i < NP; i += 128) {
        px[i] = xb[(size_t)i * 10 + 0];
        py[i] = xb[(size_t)i * 10 + 1];
    }
    __syncthreads();

    const int w    = tid >> 5;
    const int lane = tid & 31;
    const int n    = n0 + w;
    const float pxn = px[n], pyn = py[n];

    for (int c = lane; c < NP; c += 32) {
        float dx = px[c] - pxn, dy = py[c] - pyn;
        d2s[w][c] = (c == n) ? __builtin_huge_valf() : dx * dx + dy * dy;
    }
    __syncthreads();

    int* out = knn + ((size_t)b * NP + n) * KNN;
    for (int r = 0; r < KNN; ++r) {
        float best = __builtin_huge_valf();
        int   bi   = 0x7fffffff;
        for (int c = lane; c < NP; c += 32) {
            float v = d2s[w][c];
            if (v < best) { best = v; bi = c; }
        }
        #pragma unroll
        for (int off = 16; off > 0; off >>= 1) {
            float ov = __shfl_xor(best, off, 32);
            int   oi = __shfl_xor(bi,   off, 32);
            if (ov < best || (ov == best && oi < bi)) { best = ov; bi = oi; }
        }
        if (lane == 0) { out[r] = bi; d2s[w][bi] = __builtin_huge_valf(); }
        __syncthreads();
    }
}

// ---------------------------------------------------------------------------
// pos3 = (x, y, sum(x[2:10])) ; A0 cols 0..9 = x (f16), cols 19..31 zeroed
// ---------------------------------------------------------------------------
__global__ void pos_feat_kernel(const float* __restrict__ x,
                                float* __restrict__ pos3,
                                _Float16* __restrict__ A0)
{
    int i = blockIdx.x * blockDim.x + threadIdx.x;
    if (i >= NPTS) return;
    const float* xi = x + (size_t)i * 10;
    _Float16* a = A0 + (size_t)i * 32;
    float e = 0.f;
    #pragma unroll
    for (int d = 0; d < 10; ++d) {
        float v = xi[d];
        a[d] = (_Float16)v;
        if (d >= 2) e += v;
    }
    #pragma unroll
    for (int d = 19; d < 32; ++d) a[d] = (_Float16)0.f;
    pos3[i * 3 + 0] = xi[0];
    pos3[i * 3 + 1] = xi[1];
    pos3[i * 3 + 2] = e;
}

// ---------------------------------------------------------------------------
// 3x3 covariance of {self, 16 nbrs} in (x,y,energy) -> A0 cols 10..18
// ---------------------------------------------------------------------------
__global__ void cov_kernel(const float* __restrict__ pos3,
                           const int* __restrict__ knn,
                           _Float16* __restrict__ A0)
{
    int i = blockIdx.x * blockDim.x + threadIdx.x;
    if (i >= NPTS) return;
    int b = i >> 11;
    const int* nb = knn + (size_t)i * KNN;

    float px[17], py[17], pz[17];
    px[0] = pos3[i * 3 + 0]; py[0] = pos3[i * 3 + 1]; pz[0] = pos3[i * 3 + 2];
    #pragma unroll
    for (int k = 0; k < KNN; ++k) {
        int j = b * NP + nb[k];
        px[k + 1] = pos3[j * 3 + 0];
        py[k + 1] = pos3[j * 3 + 1];
        pz[k + 1] = pos3[j * 3 + 2];
    }
    float mx = 0.f, my = 0.f, mz = 0.f;
    #pragma unroll
    for (int k = 0; k < 17; ++k) { mx += px[k]; my += py[k]; mz += pz[k]; }
    const float inv = 1.f / 17.f;
    mx *= inv; my *= inv; mz *= inv;

    float c00 = 0.f, c01 = 0.f, c02 = 0.f, c11 = 0.f, c12 = 0.f, c22 = 0.f;
    #pragma unroll
    for (int k = 0; k < 17; ++k) {
        float dx = px[k] - mx, dy = py[k] - my, dz = pz[k] - mz;
        c00 += dx * dx; c01 += dx * dy; c02 += dx * dz;
        c11 += dy * dy; c12 += dy * dz; c22 += dz * dz;
    }
    c00 *= inv; c01 *= inv; c02 *= inv; c11 *= inv; c12 *= inv; c22 *= inv;

    _Float16* a = A0 + (size_t)i * 32 + 10;   // row-major 3x3, symmetric
    a[0] = (_Float16)c00; a[1] = (_Float16)c01; a[2] = (_Float16)c02;
    a[3] = (_Float16)c01; a[4] = (_Float16)c11; a[5] = (_Float16)c12;
    a[6] = (_Float16)c02; a[7] = (_Float16)c12; a[8] = (_Float16)c22;
}

// ---------------------------------------------------------------------------
// Weight convert+transpose: w (K x N) f32 -> wT (N x Kp) f16, zero pad k>=K
// ---------------------------------------------------------------------------
__global__ void wt_convert_kernel(const float* __restrict__ w,
                                  _Float16* __restrict__ wt,
                                  int K, int N, int Kp)
{
    int i = blockIdx.x * blockDim.x + threadIdx.x;
    if (i >= N * Kp) return;
    int n = i / Kp, k = i - n * Kp;
    wt[i] = (k < K) ? (_Float16)w[(size_t)k * N + n] : (_Float16)0.f;
}

// ---------------------------------------------------------------------------
// WMMA GEMM: C[M,N] f32 = A[M,Kp] f16 (row-major) x BT[N,Kp] f16 (row-major).
// Block = 64*wavesN threads; wave tile 32x64 (2x4 x v_wmma_f32_16x16x32_f16).
// All dims multiples of tile; uniform control flow, EXEC all ones at WMMA.
// ---------------------------------------------------------------------------
__global__ void gemm_f16_wmma(const _Float16* __restrict__ A,
                              const _Float16* __restrict__ BT,
                              float* __restrict__ C,
                              int M, int N, int Kp)
{
    const int lane   = threadIdx.x & 31;
    const int waveId = threadIdx.x >> 5;
    const int wm     = waveId & 1;
    const int wn     = waveId >> 1;
    const int wavesN = blockDim.x >> 6;
    const int mBase  = blockIdx.x * 64 + wm * 32;
    const int nBase  = blockIdx.y * (64 * wavesN) + wn * 64;
    const int lm     = lane & 15;
    const int hi     = lane >> 4;

    const v8f zero = {0.f, 0.f, 0.f, 0.f, 0.f, 0.f, 0.f, 0.f};
    v8f acc[2][4];
    #pragma unroll
    for (int i = 0; i < 2; ++i)
        #pragma unroll
        for (int j = 0; j < 4; ++j) acc[i][j] = zero;

    for (int k0 = 0; k0 < Kp; k0 += 32) {
        // A fragments (16x32 tile): lane<16 holds K 0-15 of row lm,
        // lane>=16 holds K 16-31 of row lm  -> two 16B loads.
        v16h afrag[2];
        #pragma unroll
        for (int i = 0; i < 2; ++i) {
            const _Float16* ap = A + (size_t)(mBase + i * 16 + lm) * Kp + k0 + hi * 16;
            v8h lo = *(const v8h*)ap;
            v8h hh = *(const v8h*)(ap + 8);
            #pragma unroll
            for (int e = 0; e < 8; ++e) { afrag[i][e] = lo[e]; afrag[i][e + 8] = hh[e]; }
        }
        // B fragments (32x16 tile from BT row-major): lane<16 = K 0-15 of
        // col lm, lane>=16 = K 16-31 of col lm -> one contiguous 32B load.
        v16h bfrag[4];
        #pragma unroll
        for (int j = 0; j < 4; ++j) {
            const _Float16* bp = BT + (size_t)(nBase + j * 16 + lm) * Kp + k0 + hi * 16;
            bfrag[j] = *(const v16h*)bp;
        }
        #pragma unroll
        for (int i = 0; i < 2; ++i)
            #pragma unroll
            for (int j = 0; j < 4; ++j)
                acc[i][j] = __builtin_amdgcn_wmma_f32_16x16x32_f16(
                    false, afrag[i], false, bfrag[j],
                    (short)0, acc[i][j], false, false);
    }

    // C/D layout: VGPR r of lane l holds row (r + 8*(l>=16)), col (l&15).
    #pragma unroll
    for (int i = 0; i < 2; ++i)
        #pragma unroll
        for (int j = 0; j < 4; ++j) {
            const int col = nBase + j * 16 + lm;
            #pragma unroll
            for (int r = 0; r < 8; ++r) {
                const int row = mBase + i * 16 + hi * 8 + r;
                C[(size_t)row * N + col] = acc[i][j][r];
            }
        }
}

// ---------------------------------------------------------------------------
// Training-mode BN column stats: scale = g*rsqrt(var+eps), shift = b - mean*scale
// ---------------------------------------------------------------------------
__global__ void col_stats_kernel(const float* __restrict__ C, int M, int N,
                                 const float* __restrict__ g,
                                 const float* __restrict__ bb,
                                 float* __restrict__ scale,
                                 float* __restrict__ shift)
{
    __shared__ float s1[256];
    __shared__ float s2[256];
    const int col = blockIdx.x;
    const int tid = threadIdx.x;
    float a = 0.f, q = 0.f;
    for (int r = tid; r < M; r += 256) {
        float v = C[(size_t)r * N + col];
        a += v; q += v * v;
    }
    s1[tid] = a; s2[tid] = q;
    __syncthreads();
    for (int off = 128; off > 0; off >>= 1) {
        if (tid < off) { s1[tid] += s1[tid + off]; s2[tid] += s2[tid + off]; }
        __syncthreads();
    }
    if (tid == 0) {
        float mean = s1[0] / (float)M;
        float var  = s2[0] / (float)M - mean * mean;
        float rs   = rsqrtf(var + 1e-5f);
        float sc   = g[col] * rs;
        scale[col] = sc;
        shift[col] = bb[col] - mean * sc;
    }
}

__global__ void bn_act_kernel(const float* __restrict__ C,
                              const float* __restrict__ scale,
                              const float* __restrict__ shift,
                              _Float16* __restrict__ H,
                              float* __restrict__ Of,
                              int M, int N, int relu)
{
    size_t i = (size_t)blockIdx.x * blockDim.x + threadIdx.x;
    if (i >= (size_t)M * N) return;
    int col = (int)(i % (size_t)N);
    float y = C[i] * scale[col] + shift[col];
    if (relu) y = fmaxf(y, 0.f);
    if (H)  H[i]  = (_Float16)y;
    if (Of) Of[i] = y;
}

// ---------------------------------------------------------------------------
// agg = relu(max over 16 neighbors of feat)  (one block per point)
// ---------------------------------------------------------------------------
__global__ void gather_max_kernel(const _Float16* __restrict__ H,
                                  const int* __restrict__ knn,
                                  _Float16* __restrict__ A, int C)
{
    __shared__ int sidx[KNN];
    const int p = blockIdx.x;
    const int b = p >> 11;
    if (threadIdx.x < KNN)
        sidx[threadIdx.x] = b * NP + knn[(size_t)p * KNN + threadIdx.x];
    __syncthreads();
    for (int c = threadIdx.x; c < C; c += blockDim.x) {
        float m = -__builtin_huge_valf();
        #pragma unroll
        for (int k = 0; k < KNN; ++k) {
            float v = (float)H[(size_t)sidx[k] * C + c];
            m = fmaxf(m, v);
        }
        m = fmaxf(m, 0.f);                 // relu(max(...))
        A[(size_t)p * C + c] = (_Float16)m;
    }
}

__global__ void zero_f16_kernel(_Float16* __restrict__ p, int n)
{
    int i = blockIdx.x * blockDim.x + threadIdx.x;
    if (i < n) p[i] = (_Float16)0.f;
}

// glob[b, c] = max over N points of H5  ->  A6 (64 x 1024, rows 8..63 zero)
__global__ void global_max_kernel(const _Float16* __restrict__ H5,
                                  _Float16* __restrict__ A6)
{
    const int b = blockIdx.x;
    for (int c = threadIdx.x; c < 1024; c += blockDim.x) {
        float m = -__builtin_huge_valf();
        for (int n = 0; n < NP; ++n) {
            float v = (float)H5[((size_t)b * NP + n) * 1024 + c];
            m = fmaxf(m, v);
        }
        A6[(size_t)b * 1024 + c] = (_Float16)m;
    }
}

// ---------------------------------------------------------------------------
extern "C" void kernel_launch(void* const* d_in, const int* in_sizes, int n_in,
                              void* d_out, int out_size, void* d_ws, size_t ws_size,
                              hipStream_t stream)
{
    (void)in_sizes; (void)n_in; (void)out_size; (void)ws_size;

    const float* x   = (const float*)d_in[0];
    const float* w1a = (const float*)d_in[1];
    const float* g1a = (const float*)d_in[2];
    const float* b1a = (const float*)d_in[3];
    const float* w1b = (const float*)d_in[4];
    const float* g1b = (const float*)d_in[5];
    const float* b1b = (const float*)d_in[6];
    const float* w1c = (const float*)d_in[7];
    const float* g1c = (const float*)d_in[8];
    const float* b1c = (const float*)d_in[9];
    const float* wg1 = (const float*)d_in[10];
    const float* gg1 = (const float*)d_in[11];
    const float* bg1 = (const float*)d_in[12];
    const float* wg2 = (const float*)d_in[13];
    const float* gg2 = (const float*)d_in[14];
    const float* bg2 = (const float*)d_in[15];
    const float* w2  = (const float*)d_in[16];
    const float* g2  = (const float*)d_in[17];
    const float* b2  = (const float*)d_in[18];

    char* p = (char*)d_ws;
    auto carve = [&](size_t bytes) -> char* {
        char* r = p;
        p += (bytes + 255) & ~(size_t)255;
        return r;
    };
    int*      idx   = (int*)     carve((size_t)NPTS * KNN * 4);
    float*    pos3  = (float*)   carve((size_t)NPTS * 3 * 4);
    _Float16* A0    = (_Float16*)carve((size_t)NPTS * 32 * 2);
    _Float16* Ha    = (_Float16*)carve((size_t)NPTS * 64 * 2);
    _Float16* Hb    = (_Float16*)carve((size_t)NPTS * 64 * 2);
    _Float16* A4    = (_Float16*)carve((size_t)NPTS * 64 * 2);
    _Float16* H4    = (_Float16*)carve((size_t)NPTS * 512 * 2);
    _Float16* A5    = (_Float16*)carve((size_t)NPTS * 512 * 2);
    _Float16* H5    = (_Float16*)carve((size_t)NPTS * 1024 * 2);
    _Float16* A6    = (_Float16*)carve((size_t)64 * 1024 * 2);
    float*    Cbuf  = (float*)   carve((size_t)NPTS * 1024 * 4);
    float*    scale = (float*)   carve(1024 * 4);
    float*    shift = (float*)   carve(1024 * 4);
    _Float16* w1aT  = (_Float16*)carve((size_t)64 * 32 * 2);
    _Float16* w1bT  = (_Float16*)carve((size_t)64 * 64 * 2);
    _Float16* w1cT  = (_Float16*)carve((size_t)64 * 64 * 2);
    _Float16* wg1T  = (_Float16*)carve((size_t)512 * 64 * 2);
    _Float16* wg2T  = (_Float16*)carve((size_t)1024 * 512 * 2);
    _Float16* w2T   = (_Float16*)carve((size_t)512 * 1024 * 2);

    // weights -> transposed f16 (B operand: contiguous K per output column)
    wt_convert_kernel<<<(64 * 32 + 255) / 256, 256, 0, stream>>>(w1a, w1aT, 19, 64, 32);
    wt_convert_kernel<<<(64 * 64 + 255) / 256, 256, 0, stream>>>(w1b, w1bT, 64, 64, 64);
    wt_convert_kernel<<<(64 * 64 + 255) / 256, 256, 0, stream>>>(w1c, w1cT, 64, 64, 64);
    wt_convert_kernel<<<(512 * 64 + 255) / 256, 256, 0, stream>>>(wg1, wg1T, 64, 512, 64);
    wt_convert_kernel<<<(1024 * 512 + 255) / 256, 256, 0, stream>>>(wg2, wg2T, 512, 1024, 512);
    wt_convert_kernel<<<(512 * 1024 + 255) / 256, 256, 0, stream>>>(w2, w2T, 1024, 512, 1024);

    // kNN + covariance features
    knn_kernel<<<BB * (NP / 4), 128, 0, stream>>>(x, idx);
    pos_feat_kernel<<<NPTS / 256, 256, 0, stream>>>(x, pos3, A0);
    cov_kernel<<<NPTS / 256, 256, 0, stream>>>(pos3, idx, A0);

    // MLP layer 1a: 19(pad32) -> 64, BN+ReLU
    gemm_f16_wmma<<<dim3(NPTS / 64, 1), 64, 0, stream>>>(A0, w1aT, Cbuf, NPTS, 64, 32);
    col_stats_kernel<<<64, 256, 0, stream>>>(Cbuf, NPTS, 64, g1a, b1a, scale, shift);
    bn_act_kernel<<<(NPTS * 64 + 255) / 256, 256, 0, stream>>>(Cbuf, scale, shift, Ha, nullptr, NPTS, 64, 1);
    // 1b
    gemm_f16_wmma<<<dim3(NPTS / 64, 1), 64, 0, stream>>>(Ha, w1bT, Cbuf, NPTS, 64, 64);
    col_stats_kernel<<<64, 256, 0, stream>>>(Cbuf, NPTS, 64, g1b, b1b, scale, shift);
    bn_act_kernel<<<(NPTS * 64 + 255) / 256, 256, 0, stream>>>(Cbuf, scale, shift, Hb, nullptr, NPTS, 64, 1);
    // 1c
    gemm_f16_wmma<<<dim3(NPTS / 64, 1), 64, 0, stream>>>(Hb, w1cT, Cbuf, NPTS, 64, 64);
    col_stats_kernel<<<64, 256, 0, stream>>>(Cbuf, NPTS, 64, g1c, b1c, scale, shift);
    bn_act_kernel<<<(NPTS * 64 + 255) / 256, 256, 0, stream>>>(Cbuf, scale, shift, Ha, nullptr, NPTS, 64, 1);

    // graph max pool 1: relu(max_nbr) -> GEMM 64->512 -> BN (no relu)
    gather_max_kernel<<<NPTS, 128, 0, stream>>>(Ha, idx, A4, 64);
    gemm_f16_wmma<<<dim3(NPTS / 64, 512 / 128), 128, 0, stream>>>(A4, wg1T, Cbuf, NPTS, 512, 64);
    col_stats_kernel<<<512, 256, 0, stream>>>(Cbuf, NPTS, 512, gg1, bg1, scale, shift);
    bn_act_kernel<<<(NPTS * 512 + 255) / 256, 256, 0, stream>>>(Cbuf, scale, shift, H4, nullptr, NPTS, 512, 0);

    // graph max pool 2: relu(max_nbr) -> GEMM 512->1024 -> BN (no relu)
    gather_max_kernel<<<NPTS, 256, 0, stream>>>(H4, idx, A5, 512);
    gemm_f16_wmma<<<dim3(NPTS / 64, 1024 / 128), 128, 0, stream>>>(A5, wg2T, Cbuf, NPTS, 1024, 512);
    col_stats_kernel<<<1024, 256, 0, stream>>>(Cbuf, NPTS, 1024, gg2, bg2, scale, shift);
    bn_act_kernel<<<(NPTS * 1024 + 255) / 256, 256, 0, stream>>>(Cbuf, scale, shift, H5, nullptr, NPTS, 1024, 0);

    // global max over points -> (8,1024), zero-padded to 64 rows for WMMA
    zero_f16_kernel<<<(64 * 1024 + 255) / 256, 256, 0, stream>>>(A6, 64 * 1024);
    global_max_kernel<<<BB, 256, 0, stream>>>(H5, A6);

    // final: (64x1024) x (1024->512) -> BN over true 8 rows, no relu, f32 out
    gemm_f16_wmma<<<dim3(1, 512 / 128), 128, 0, stream>>>(A6, w2T, Cbuf, 64, 512, 1024);
    col_stats_kernel<<<512, 256, 0, stream>>>(Cbuf, 8, 512, g2, b2, scale, shift);
    bn_act_kernel<<<(8 * 512 + 255) / 256, 256, 0, stream>>>(Cbuf, scale, shift, nullptr, (float*)d_out, 8, 512, 0);
}
